// ChemistryAwareDecoder_62912680952405
// MI455X (gfx1250) — matrix operand order repaired and checked by reading, not compiled, not run on hardware
//
#include <hip/hip_runtime.h>
#include <hip/hip_bf16.h>

typedef __attribute__((ext_vector_type(16))) __bf16 v16bf;
typedef __attribute__((ext_vector_type(4)))  __bf16 v4bf;
typedef __attribute__((ext_vector_type(8)))  float  v8f;

#define MT 16              // edges per wave tile (WMMA M)
#define SW_KT 4
#define SW_NT 4
#define CW_KT 24
#define CW_NT 12
#define MW_KT 28
#define MW_NT 8
#define SW_ELEMS (SW_KT*SW_NT*512)   /* 8192   */
#define CW_ELEMS (CW_KT*CW_NT*512)   /* 147456 */
#define MW_ELEMS (MW_KT*MW_NT*512)   /* 114688 */
#define TOTAL_W  (SW_ELEMS+CW_ELEMS+MW_ELEMS)

// Pack 4 products for 4-aligned column group k0..k0+3 of row m into the
// 16x32 bf16 A-fragment layout (ISA 7.12.2): the 4 elements occupy 8
// consecutive, 8-byte-aligned bytes of one lane-row -> single ds_store_b64.
__device__ __forceinline__ void store_pack4(unsigned char* ldsA, int m, int k0,
                                            float a, float b, float c, float d) {
  int kt = k0 >> 5, kk = k0 & 31;
  int h = (kk >> 3) & 1;
  int v0 = (kk < 16) ? ((kk & 7) >> 1) : (4 + ((kk & 7) >> 1));
  v4bf pk = {(__bf16)a, (__bf16)b, (__bf16)c, (__bf16)d};
  *(v4bf*)(ldsA + kt * 1024 + (m + 16 * h) * 32 + 4 * v0) = pk;
}

// Pre-pack f32 row-major weights (K x N) into bf16 WMMA B-fragments:
// frag order [nt][kt][lane][slot], lane L holds N = nt*16 + (L%16),
// K = kt*32 + (slot < 8 ? 8*(L/16) + slot : 16 + 8*(L/16) + slot - 8).
__global__ void prep_weights_kernel(const float* __restrict__ sw1,
                                    const float* __restrict__ cw1,
                                    const float* __restrict__ mw1,
                                    __bf16* __restrict__ ws) {
  int tid = blockIdx.x * blockDim.x + threadIdx.x;
  const float* W;
  __bf16* out;
  int KT, Nout, f;
  if (tid < SW_ELEMS) {
    W = sw1; out = ws; KT = SW_KT; Nout = 64; f = tid;
  } else if (tid < SW_ELEMS + CW_ELEMS) {
    W = cw1; out = ws + SW_ELEMS; KT = CW_KT; Nout = 192; f = tid - SW_ELEMS;
  } else if (tid < TOTAL_W) {
    W = mw1; out = ws + SW_ELEMS + CW_ELEMS; KT = MW_KT; Nout = 128;
    f = tid - SW_ELEMS - CW_ELEMS;
  } else {
    return;
  }
  int slot = f & 15;
  int lane = (f >> 4) & 31;
  int rem  = f >> 9;
  int kt = rem % KT;
  int nt = rem / KT;
  int h = lane >> 4;
  int K = kt * 32 + ((slot < 8) ? (8 * h + slot) : (16 + 8 * h + (slot - 8)));
  int Nn = nt * 16 + (lane & 15);
  out[f] = (__bf16)W[K * Nout + Nn];
}

// One GEMM path: A from LDS frags [aTile0 .. aTile0+KT), B from pre-packed ws.
// Epilogue: +bias, relu, * w2[n], reduce over n, + b2 -> outScore[16] (LDS).
__device__ __forceinline__ void gemm_path(const unsigned char* ldsA, int aTile0,
                                          int KT, int NT,
                                          const __bf16* __restrict__ B,
                                          const float* __restrict__ b1,
                                          const float* __restrict__ w2,
                                          float b2, float* outScore, int lane) {
  float s[8] = {0.f, 0.f, 0.f, 0.f, 0.f, 0.f, 0.f, 0.f};
  for (int nt = 0; nt < NT; ++nt) {
    v8f acc = {0.f, 0.f, 0.f, 0.f, 0.f, 0.f, 0.f, 0.f};
#pragma unroll 2
    for (int kt = 0; kt < KT; ++kt) {
      v16bf a = *(const v16bf*)(ldsA + (size_t)(aTile0 + kt) * 1024 + lane * 32);
      v16bf b = *(const v16bf*)(B + ((size_t)(nt * KT + kt) * 32 + lane) * 16);
      acc = __builtin_amdgcn_wmma_f32_16x16x32_bf16(
          false, a, false, b, (short)0, acc, false, false);
    }
    int n = nt * 16 + (lane & 15);
    float bias = b1[n];
    float w = w2[n];
#pragma unroll
    for (int i = 0; i < 8; ++i) {
      float v = acc[i] + bias;
      v = v > 0.f ? v : 0.f;
      s[i] += v * w;
    }
  }
  // reduce over n: lanes 0-15 share rows m=0..7, lanes 16-31 rows m=8..15
#pragma unroll
  for (int off = 1; off < 16; off <<= 1) {
#pragma unroll
    for (int i = 0; i < 8; ++i) s[i] += __shfl_xor(s[i], off, 32);
  }
  if ((lane & 15) == 0) {
    int mbase = (lane >> 4) * 8;
#pragma unroll
    for (int i = 0; i < 8; ++i) outScore[mbase + i] = s[i] + b2;
  }
}

__global__ __launch_bounds__(32) void edge_score_kernel(
    const float* __restrict__ z, const float* __restrict__ chem,
    const int* __restrict__ edge, const int* __restrict__ mask,
    const float* __restrict__ sb1, const float* __restrict__ sw2,
    const float* __restrict__ sb2, const float* __restrict__ cb1,
    const float* __restrict__ cw2, const float* __restrict__ cb2,
    const float* __restrict__ mb1, const float* __restrict__ mw2,
    const float* __restrict__ mb2, const float* __restrict__ pw,
    const __bf16* __restrict__ ws, float* __restrict__ out, int E) {
  __shared__ __align__(32) unsigned char ldsA[28 * 1024];  // 28 K-tiles of 16x32 bf16
  __shared__ float scoreS[MT], scoreC[MT], scoreM[MT];
  __shared__ int srcs[MT], dsts[MT];

  const int lane = threadIdx.x;
  const int ebase = blockIdx.x * MT;

  if (lane < MT) {
    int ce = ebase + lane;
    if (ce >= E) ce = E - 1;   // clamp tail (grid is exact for E=200000)
    srcs[lane] = edge[2 * ce];
    dsts[lane] = edge[2 * ce + 1];
  }
  __syncthreads();

  // Gather node rows, form elementwise products, pack bf16 into A-frag layout.
  // Each lane owns a 4-aligned column group -> one packed b64 LDS store each.
  for (int m = 0; m < MT; ++m) {
    const int si = srcs[m];
    const int di = dsts[m];
    {  // structural: cols 0..127 (A tiles 0..3)
      const float4 zs = *(const float4*)(z + (size_t)si * 128 + 4 * lane);
      const float4 zd = *(const float4*)(z + (size_t)di * 128 + 4 * lane);
      store_pack4(ldsA, m, 4 * lane,
                  zs.x * zd.x, zs.y * zd.y, zs.z * zd.z, zs.w * zd.w);
    }
#pragma unroll
    for (int it = 0; it < 6; ++it) {  // chemical: cols 128..895 (A tiles 4..27)
      const float4 cs = *(const float4*)(chem + (size_t)si * 768 + it * 128 + 4 * lane);
      const float4 cd = *(const float4*)(chem + (size_t)di * 768 + it * 128 + 4 * lane);
      store_pack4(ldsA, m, 128 + it * 128 + 4 * lane,
                  cs.x * cd.x, cs.y * cd.y, cs.z * cd.z, cs.w * cd.w);
    }
  }
  __syncthreads();

  // Three GEMM paths (all lanes active -> EXEC all-1s for WMMA).
  gemm_path(ldsA, 0, SW_KT, SW_NT, ws, sb1, sw2, sb2[0], scoreS, lane);
  gemm_path(ldsA, 4, CW_KT, CW_NT, ws + SW_ELEMS, cb1, cw2, cb2[0], scoreC, lane);
  gemm_path(ldsA, 0, MW_KT, MW_NT, ws + SW_ELEMS + CW_ELEMS, mb1, mw2, mb2[0],
            scoreM, lane);
  __syncthreads();

  if (lane < MT && (ebase + lane) < E) {
    float w0 = pw[0], w1 = pw[1], w2v = pw[2];
    float mx = fmaxf(w0, fmaxf(w1, w2v));
    float x0 = __expf(w0 - mx), x1 = __expf(w1 - mx), x2 = __expf(w2v - mx);
    float inv = 1.f / (x0 + x1 + x2);
    float sS = scoreS[lane], sC = scoreC[lane], sM = scoreM[lane];
    int bs = mask[srcs[lane]];
    int bd = mask[dsts[lane]];
    float res = (bs && bd) ? (x0 * sS + x1 * sC + x2 * sM) * inv : sS;
    out[ebase + lane] = res;
  }
}

extern "C" void kernel_launch(void* const* d_in, const int* in_sizes, int n_in,
                              void* d_out, int out_size, void* d_ws, size_t ws_size,
                              hipStream_t stream) {
  const float* z    = (const float*)d_in[0];
  const float* chem = (const float*)d_in[1];
  const int*   edge = (const int*)d_in[2];
  const int*   mask = (const int*)d_in[3];
  const float* sw1  = (const float*)d_in[4];
  const float* sb1  = (const float*)d_in[5];
  const float* sw2  = (const float*)d_in[6];
  const float* sb2  = (const float*)d_in[7];
  const float* cw1  = (const float*)d_in[8];
  const float* cb1  = (const float*)d_in[9];
  const float* cw2  = (const float*)d_in[10];
  const float* cb2  = (const float*)d_in[11];
  const float* mw1  = (const float*)d_in[12];
  const float* mb1  = (const float*)d_in[13];
  const float* mw2  = (const float*)d_in[14];
  const float* mb2  = (const float*)d_in[15];
  const float* pw   = (const float*)d_in[16];
  __bf16* ws = (__bf16*)d_ws;

  const int E = out_size;  // 200000

  // One-shot weight repack into bf16 WMMA B-fragments (idempotent, graph-safe).
  prep_weights_kernel<<<(TOTAL_W + 255) / 256, 256, 0, stream>>>(sw1, cw1, mw1, ws);

  // One wave32 per 16-edge tile.
  edge_score_kernel<<<(E + MT - 1) / MT, 32, 0, stream>>>(
      z, chem, edge, mask, sb1, sw2, sb2, cb1, cw2, cb2, mb1, mw2, mb2, pw, ws,
      (float*)d_out, E);
}
